// MultiHeadAttention_22162031247702
// MI455X (gfx1250) — matrix operand order repaired and checked by reading, not compiled, not run on hardware
//
#include <hip/hip_runtime.h>

// ---- problem dims (fixed by the reference) ----
#define BB 4
#define NN 2048
#define DD 1024
#define HH 16
#define EE 64
#define D3 3072
#define BN 8192   // B*N

typedef __bf16 bf16_t;
typedef __attribute__((ext_vector_type(16))) __bf16 bf16x16;
typedef __attribute__((ext_vector_type(8)))  float  f32x8;
typedef __attribute__((ext_vector_type(4)))  int    i32x4;

// CDNA5 async global->LDS path (gated: falls back to ds_store staging if absent)
// Probe round 2 showed the builtin exists with params: (v4i32 AS1*, v4i32 AS3*, Iimm, Iimm)
#if defined(__AMDGCN__) && __has_builtin(__builtin_amdgcn_global_load_async_to_lds_b128)
#define USE_ASYNC_LDS 1
typedef i32x4 __attribute__((address_space(1)))* gp1;   // global (device) v4i32*
typedef i32x4 __attribute__((address_space(3)))* lp3;   // LDS (shared) v4i32*
#else
#define USE_ASYNC_LDS 0
#endif

static __device__ __forceinline__ void wait_async_zero() {
#if USE_ASYNC_LDS
#if __has_builtin(__builtin_amdgcn_s_wait_asynccnt)
  __builtin_amdgcn_s_wait_asynccnt(0);
#else
  asm volatile("s_wait_asynccnt 0" ::: "memory");
#endif
#endif
}

static __device__ __forceinline__ bf16_t tobf(float f) {
  // round-to-nearest-even fp32 -> bf16 without relying on cvt builtins
  unsigned u = __builtin_bit_cast(unsigned, f);
  u += 0x7FFFu + ((u >> 16) & 1u);
  unsigned short hs = (unsigned short)(u >> 16);
  return __builtin_bit_cast(bf16_t, hs);
}

union FragU { bf16x16 v; float4 q[2]; };

// ---------------- conversion kernels ----------------
__global__ void cvt_bf16_kernel(const float* __restrict__ in, bf16_t* __restrict__ out, int n) {
  int stride = gridDim.x * blockDim.x;
  for (int i = blockIdx.x * blockDim.x + threadIdx.x; i < n; i += stride)
    out[i] = tobf(in[i]);
}

// out[c][r] = bf16(in[r][c]) : produces W^T in bf16 (contiguous K rows for WMMA B)
__global__ void cvt_transpose_kernel(const float* __restrict__ in, bf16_t* __restrict__ out,
                                     int rows, int cols) {
  int n = rows * cols;
  int stride = gridDim.x * blockDim.x;
  for (int i = blockIdx.x * blockDim.x + threadIdx.x; i < n; i += stride) {
    int r = i / cols, c = i - r * cols;
    out[(size_t)c * rows + r] = tobf(in[i]);
  }
}

// ---------------- bf16 WMMA GEMM ----------------
// C[M,Ncols] = A[M,K] * B[K,Ncols], with B provided transposed (BT[Ncols][K], row-major).
// Block tile 128x128, 8 waves each computing 32x64 (2x4 tiles of 16x16), K-step 32.
// Staging: double-buffered GLOBAL_LOAD_ASYNC_TO_LDS_B128 pipeline when available,
// else synchronous global->reg->ds_store.
// MODE 0: scatter epilogue into Q / K / V^T (bf16).  MODE 1: fp32 row-major output.
template <int MODE>
__global__ __launch_bounds__(256) void gemm_bf16_wmma(
    const bf16_t* __restrict__ A, const bf16_t* __restrict__ BT, int Kdim,
    float* __restrict__ outF,
    bf16_t* __restrict__ outQ, bf16_t* __restrict__ outK, bf16_t* __restrict__ outVT)
{
  constexpr int LDS = 40;  // 32 bf16 + 8 pad -> 80B row stride, 16B aligned
#if USE_ASYNC_LDS
  __shared__ __align__(16) bf16_t sA[2][128 * LDS];
  __shared__ __align__(16) bf16_t sB[2][128 * LDS];
#else
  __shared__ __align__(16) bf16_t sA[1][128 * LDS];
  __shared__ __align__(16) bf16_t sB[1][128 * LDS];
#endif

  const int tid  = threadIdx.x;
  const int wave = tid >> 5;
  const int lane = tid & 31;
  const int r    = lane & 15;
  const int half = lane >> 4;
  const int m0 = blockIdx.x * 128;
  const int n0 = blockIdx.y * 128;
  const int wr = (wave & 3) * 32;   // wave row offset in block tile
  const int wc = (wave >> 2) * 64;  // wave col offset in block tile

  f32x8 acc[2][4] = {};

#if USE_ASYNC_LDS
  // issue one staging wave of async copies: 2 x 16B per thread per matrix
  auto stage = [&](int buf, int k0) {
    for (int c = tid; c < 512; c += 256) {
      int row = c >> 2, seg = c & 3;
      __builtin_amdgcn_global_load_async_to_lds_b128(
          (gp1)(uintptr_t)&A[(size_t)(m0 + row) * Kdim + k0 + seg * 8],
          (lp3)(unsigned)(uintptr_t)&sA[buf][row * LDS + seg * 8], 0, 0);
      __builtin_amdgcn_global_load_async_to_lds_b128(
          (gp1)(uintptr_t)&BT[(size_t)(n0 + row) * Kdim + k0 + seg * 8],
          (lp3)(unsigned)(uintptr_t)&sB[buf][row * LDS + seg * 8], 0, 0);
    }
  };
  stage(0, 0);
#endif

  for (int k0 = 0; k0 < Kdim; k0 += 32) {
#if USE_ASYNC_LDS
    const int buf = (k0 >> 5) & 1;
    wait_async_zero();     // my async copies into `buf` have landed in LDS
    __syncthreads();       // everyone's have; everyone also done reading buf^1
    if (k0 + 32 < Kdim) stage(buf ^ 1, k0 + 32);  // overlap next stage with compute
#else
    const int buf = 0;
    for (int c = tid; c < 512; c += 256) {
      int row = c >> 2, seg = c & 3;
      *reinterpret_cast<float4*>(&sA[0][row * LDS + seg * 8]) =
          *reinterpret_cast<const float4*>(&A[(size_t)(m0 + row) * Kdim + k0 + seg * 8]);
      *reinterpret_cast<float4*>(&sB[0][row * LDS + seg * 8]) =
          *reinterpret_cast<const float4*>(&BT[(size_t)(n0 + row) * Kdim + k0 + seg * 8]);
    }
    __syncthreads();
    if (k0 + 32 < Kdim) {  // hint next K tile -> global_prefetch_b8
      __builtin_prefetch(&A[(size_t)(m0 + (tid >> 1)) * Kdim + k0 + 32], 0, 1);
      __builtin_prefetch(&BT[(size_t)(n0 + (tid >> 1)) * Kdim + k0 + 32], 0, 1);
    }
#endif

    // A fragments (16x32 bf16): lane<16 holds M=lane, K in [0,8)+[16,24); lane>=16: +8
    bf16x16 af[2];
    #pragma unroll
    for (int i = 0; i < 2; ++i) {
      FragU u;
      const bf16_t* p = &sA[buf][(wr + i * 16 + r) * LDS + half * 8];
      u.q[0] = *reinterpret_cast<const float4*>(p);
      u.q[1] = *reinterpret_cast<const float4*>(p + 16);
      af[i] = u.v;
    }
    // B fragments (32x16 bf16): lane holds column (lane&15), K in [0,8)+[16,24); lane>=16: +8
    bf16x16 bq[4];
    #pragma unroll
    for (int j = 0; j < 4; ++j) {
      FragU u;
      const bf16_t* p = &sB[buf][(wc + j * 16 + r) * LDS + half * 8];
      u.q[0] = *reinterpret_cast<const float4*>(p);
      u.q[1] = *reinterpret_cast<const float4*>(p + 16);
      bq[j] = u.v;
    }
    #pragma unroll
    for (int i = 0; i < 2; ++i)
      #pragma unroll
      for (int j = 0; j < 4; ++j)
        acc[i][j] = __builtin_amdgcn_wmma_f32_16x16x32_bf16(
            false, af[i], false, bq[j], (short)0, acc[i][j], false, false);
#if !USE_ASYNC_LDS
    __syncthreads();
#endif
  }

  // epilogue: C/D layout is (M = vr + 8*half, N = lane&15) for VGPR vr
  #pragma unroll
  for (int i = 0; i < 2; ++i) {
    #pragma unroll
    for (int j = 0; j < 4; ++j) {
      #pragma unroll
      for (int vr = 0; vr < 8; ++vr) {
        int gm = m0 + wr + i * 16 + vr + 8 * half;
        int gn = n0 + wc + j * 16 + r;
        float val = acc[i][j][vr];
        if (MODE == 0) {
          int part = gn >> 10;        // 0=Q, 1=K, 2=V
          int cc = gn & 1023;
          int hh = cc >> 6;           // head
          int ee = cc & 63;           // within-head dim
          int bb = gm >> 11;          // batch (N=2048)
          int nn = gm & 2047;         // sequence position
          if (part == 0)
            outQ[((size_t)(bb * HH + hh) * NN + nn) * EE + ee] = tobf(val);
          else if (part == 1)
            outK[((size_t)(bb * HH + hh) * NN + nn) * EE + ee] = tobf(val);
          else  // store V transposed: [b,h,e,n] so PV WMMA B-operand is contiguous in j
            outVT[((size_t)(bb * HH + hh) * EE + ee) * NN + nn] = tobf(val);
        } else {
          outF[(size_t)gm * DD + gn] = val;
        }
      }
    }
  }
}

// ---------------- flash-style attention ----------------
// scores[i,j] = k_i . q_j / 8, masked to j >= i, softmax over j, out[i] = sum_j p * v_j.
// One wave owns 16 rows i; streams j in chunks of 32 with online softmax.
__global__ __launch_bounds__(128) void attn_fwd_wmma(
    const bf16_t* __restrict__ Q, const bf16_t* __restrict__ K,
    const bf16_t* __restrict__ VT, bf16_t* __restrict__ attn)
{
  __shared__ __align__(16) bf16_t sP[4][16 * 32];  // per-wave P staging (A-fragment relayout)

  const int tid  = threadIdx.x;
  const int wave = tid >> 5;
  const int lane = tid & 31;
  const int r    = lane & 15;
  const int half = lane >> 4;
  const int bh = blockIdx.x;           // b*H + h
  const int b  = bh >> 4;
  const int h  = bh & 15;
  const int i0 = (blockIdx.y * 4 + wave) * 16;

  const bf16_t* Qh = Q  + (size_t)bh * NN * EE;
  const bf16_t* Kh = K  + (size_t)bh * NN * EE;
  const bf16_t* Vh = VT + (size_t)bh * EE * NN;
  bf16_t* myP = sP[wave];

  // A-fragments of K rows i0..i0+15 over E=64 (two K-chunks of 32), loaded once
  bf16x16 aK[2];
  #pragma unroll
  for (int ec = 0; ec < 2; ++ec) {
    FragU u;
    const bf16_t* p = Kh + (size_t)(i0 + r) * EE + ec * 32 + half * 8;
    u.q[0] = *reinterpret_cast<const float4*>(p);
    u.q[1] = *reinterpret_cast<const float4*>(p + 16);
    aK[ec] = u.v;
  }

  f32x8 o[4] = {};              // 16 x 64 output accumulator (4 N-tiles)
  float mrow[8], lrow[8];
  #pragma unroll
  for (int vr = 0; vr < 8; ++vr) { mrow[vr] = -1e30f; lrow[vr] = 0.0f; }

  for (int j0 = i0 & ~31; j0 < NN; j0 += 32) {
    f32x8 s[2];
    #pragma unroll
    for (int t = 0; t < 2; ++t) {
      int jt = j0 + t * 16;
      f32x8 sc = {};
      #pragma unroll
      for (int ec = 0; ec < 2; ++ec) {  // B = Q^T chunk (32e x 16j): Q row-major is col-major here
        FragU u;
        const bf16_t* p = Qh + (size_t)(jt + r) * EE + ec * 32 + half * 8;
        u.q[0] = *reinterpret_cast<const float4*>(p);
        u.q[1] = *reinterpret_cast<const float4*>(p + 16);
        sc = __builtin_amdgcn_wmma_f32_16x16x32_bf16(
            false, aK[ec], false, u.v, (short)0, sc, false, false);
      }
      #pragma unroll
      for (int vr = 0; vr < 8; ++vr) {
        int gi = i0 + vr + 8 * half;   // row i (key pos)
        int gj = jt + r;               // col j (query pos)
        float v = sc[vr] * 0.125f;     // 1/sqrt(64)
        s[t][vr] = (gj < gi) ? -1e30f : v;   // keep i <= j
      }
    }

    // online softmax over the 32-j chunk; row reductions across 16-lane halves
    float fac[8];
    #pragma unroll
    for (int vr = 0; vr < 8; ++vr) {
      float mt = fmaxf(s[0][vr], s[1][vr]);
      mt = fmaxf(mt, __shfl_xor(mt, 1));
      mt = fmaxf(mt, __shfl_xor(mt, 2));
      mt = fmaxf(mt, __shfl_xor(mt, 4));
      mt = fmaxf(mt, __shfl_xor(mt, 8));
      float mnew = fmaxf(mrow[vr], mt);
      float f = __expf(mrow[vr] - mnew);
      fac[vr] = f;
      mrow[vr] = mnew;
      float p0 = __expf(s[0][vr] - mnew);
      float p1 = __expf(s[1][vr] - mnew);
      s[0][vr] = p0; s[1][vr] = p1;
      float sum = p0 + p1;
      sum += __shfl_xor(sum, 1);
      sum += __shfl_xor(sum, 2);
      sum += __shfl_xor(sum, 4);
      sum += __shfl_xor(sum, 8);
      lrow[vr] = lrow[vr] * f + sum;
    }
    #pragma unroll
    for (int et = 0; et < 4; ++et)
      #pragma unroll
      for (int vr = 0; vr < 8; ++vr)
        o[et][vr] *= fac[vr];

    // P (16x32) -> LDS, then reload as WMMA A-fragment
    #pragma unroll
    for (int t = 0; t < 2; ++t)
      #pragma unroll
      for (int vr = 0; vr < 8; ++vr)
        myP[(vr + 8 * half) * 32 + t * 16 + r] = tobf(s[t][vr]);

    FragU up;
    const bf16_t* pp = &myP[r * 32 + half * 8];
    up.q[0] = *reinterpret_cast<const float4*>(pp);
    up.q[1] = *reinterpret_cast<const float4*>(pp + 16);

    // out += P(16x32) x V^T-chunk(32j x 16e), 4 e-tiles
    #pragma unroll
    for (int et = 0; et < 4; ++et) {
      FragU uv;
      const bf16_t* vp = Vh + (size_t)(et * 16 + r) * NN + j0 + half * 8;
      uv.q[0] = *reinterpret_cast<const float4*>(vp);
      uv.q[1] = *reinterpret_cast<const float4*>(vp + 16);
      o[et] = __builtin_amdgcn_wmma_f32_16x16x32_bf16(
          false, up.v, false, uv.v, (short)0, o[et], false, false);
    }
  }

  // write attn output as [b, n, h, e] row-major (= [BN, D]) in bf16 for the out-proj GEMM
  #pragma unroll
  for (int et = 0; et < 4; ++et) {
    #pragma unroll
    for (int vr = 0; vr < 8; ++vr) {
      int gi  = i0 + vr + 8 * half;
      int col = h * EE + et * 16 + r;
      float val = o[et][vr] / lrow[vr];
      attn[(size_t)(b * NN + gi) * DD + col] = tobf(val);
    }
  }
}

// ---------------- launch ----------------
extern "C" void kernel_launch(void* const* d_in, const int* in_sizes, int n_in,
                              void* d_out, int out_size, void* d_ws, size_t ws_size,
                              hipStream_t stream)
{
  (void)in_sizes; (void)n_in; (void)out_size; (void)ws_size;
  const float* x    = (const float*)d_in[0];  // [B,N,D]
  const float* Wqkv = (const float*)d_in[1];  // [D,3D]
  const float* Wo   = (const float*)d_in[2];  // [D,D]
  float* out = (float*)d_out;                 // [B,N,D]

  // workspace layout (72 MB total); Xb region is reused for the attention output
  char* w = (char*)d_ws;
  const size_t MB = 1024 * 1024;
  bf16_t* Xb    = (bf16_t*)(w);             // 16 MB : x in bf16, later attn output
  bf16_t* WqkvT = (bf16_t*)(w + 16 * MB);   //  6 MB
  bf16_t* WoT   = (bf16_t*)(w + 22 * MB);   //  2 MB
  bf16_t* Qb    = (bf16_t*)(w + 24 * MB);   // 16 MB : [B,H,N,E]
  bf16_t* Kb    = (bf16_t*)(w + 40 * MB);   // 16 MB : [B,H,N,E]
  bf16_t* VTb   = (bf16_t*)(w + 56 * MB);   // 16 MB : [B,H,E,N]
  bf16_t* attnB = Xb;

  cvt_bf16_kernel<<<4096, 256, 0, stream>>>(x, Xb, BN * DD);
  cvt_transpose_kernel<<<4096, 256, 0, stream>>>(Wqkv, WqkvT, DD, D3);
  cvt_transpose_kernel<<<2048, 256, 0, stream>>>(Wo, WoT, DD, DD);

  // QKV projection: [8192,1024] x [1024,3072] -> scatter to Q/K/V^T
  gemm_bf16_wmma<0><<<dim3(BN / 128, D3 / 128), 256, 0, stream>>>(
      Xb, WqkvT, DD, nullptr, Qb, Kb, VTb);

  // attention: 64 (b,h) pairs x 32 blocks of 4 row-tiles
  attn_fwd_wmma<<<dim3(BB * HH, NN / 64), 128, 0, stream>>>(Qb, Kb, VTb, attnB);

  // output projection: [8192,1024] x [1024,1024] -> fp32 out
  gemm_bf16_wmma<1><<<dim3(BN / 128, DD / 128), 256, 0, stream>>>(
      attnB, WoT, DD, out, nullptr, nullptr, nullptr);
}
